// CustomRNN_44959717655238
// MI455X (gfx1250) — matrix operand order repaired and checked by reading, not compiled, not run on hardware
//
#include <hip/hip_runtime.h>

// ---------------------------------------------------------------------------
// 2-layer tanh RNN for MI455X (gfx1250, wave32, WMMA).
//   Layer l:  XW = act @ W_ih^T + (b_ih + b_hh)      (big GEMM, bf16 WMMA)
//             persistent scan kernel: 512 steps of
//                 h = tanh(XW[:,t,:] + h @ W_hh^T)
//             with W_hh cached in LDS (operand-swizzled) and a device-wide
//             barrier between steps (replaces 512 kernel launches).
// ---------------------------------------------------------------------------

typedef __bf16 bf16;
typedef __attribute__((ext_vector_type(16))) __bf16 v16bf;
typedef __attribute__((ext_vector_type(8)))  __bf16 v8bf;
typedef __attribute__((ext_vector_type(8)))  float  v8f;

union ABf {
    v16bf v;
    v8bf  h[2];
};

static constexpr int BB = 64;    // batch
static constexpr int TT = 512;   // time steps
static constexpr int HH = 1024;  // hidden (== input dim)

// ---------------------------------------------------------------------------
// Elementwise helpers
// ---------------------------------------------------------------------------
__global__ void cvt_f32_to_bf16(const float* __restrict__ src,
                                bf16* __restrict__ dst, int n) {
    int i = blockIdx.x * blockDim.x + threadIdx.x;
    int stride = gridDim.x * blockDim.x;
    for (; i < n; i += stride) dst[i] = (bf16)src[i];
}

__global__ void bias_sum(const float* __restrict__ a,
                         const float* __restrict__ b,
                         float* __restrict__ o, int n) {
    int i = blockIdx.x * blockDim.x + threadIdx.x;
    if (i < n) o[i] = a[i] + b[i];
}

// ---------------------------------------------------------------------------
// Input-projection GEMM:  C[M,N] = A[M,K](bf16) * W[N,K](bf16)^T + bias[N]
// Block = 256 threads = 8 waves (2 in M x 4 in N); wave tile = 32M x 64N.
// ---------------------------------------------------------------------------
__global__ __launch_bounds__(256)
void gemm_xw(const bf16* __restrict__ A, const bf16* __restrict__ W,
             const float* __restrict__ bias, float* __restrict__ C,
             int M, int N, int K) {
    const int lane = threadIdx.x & 31;
    const int wave = threadIdx.x >> 5;      // 0..7
    const int wm   = wave & 1;              // 2 waves along M
    const int wn   = wave >> 1;             // 4 waves along N
    const int waveM = blockIdx.y * 64 + wm * 32;
    const int waveN = blockIdx.x * 256 + wn * 64;

    const int mrow = lane & 15;
    const int ksel = lane >> 4;             // 0 or 1

    v8f acc[2][4] = {};

    const bf16* A0 = A + (size_t)(waveM + mrow) * K + ksel * 8;
    const bf16* A1 = A + (size_t)(waveM + 16 + mrow) * K + ksel * 8;
    const bf16* Wb[4];
#pragma unroll
    for (int nt = 0; nt < 4; ++nt)
        Wb[nt] = W + (size_t)(waveN + nt * 16 + mrow) * K + ksel * 8;

    for (int k0 = 0; k0 < K; k0 += 32) {
        ABf a0, a1;
        a0.h[0] = *(const v8bf*)(A0 + k0);
        a0.h[1] = *(const v8bf*)(A0 + k0 + 16);
        a1.h[0] = *(const v8bf*)(A1 + k0);
        a1.h[1] = *(const v8bf*)(A1 + k0 + 16);
#pragma unroll
        for (int nt = 0; nt < 4; ++nt) {
            ABf bm;
            bm.h[0] = *(const v8bf*)(Wb[nt] + k0);
            bm.h[1] = *(const v8bf*)(Wb[nt] + k0 + 16);
            acc[0][nt] = __builtin_amdgcn_wmma_f32_16x16x32_bf16(
                false, a0.v, false, bm.v, (short)0, acc[0][nt], false, false);
            acc[1][nt] = __builtin_amdgcn_wmma_f32_16x16x32_bf16(
                false, a1.v, false, bm.v, (short)0, acc[1][nt], false, false);
        }
    }

    const int col0 = lane & 15;
    const int rb   = ksel * 8;
#pragma unroll
    for (int mt = 0; mt < 2; ++mt) {
#pragma unroll
        for (int nt = 0; nt < 4; ++nt) {
            const int col = waveN + nt * 16 + col0;
            const float bv = bias[col];
#pragma unroll
            for (int r = 0; r < 8; ++r) {
                const int row = waveM + mt * 16 + rb + r;
                C[(size_t)row * N + col] = acc[mt][nt][r] + bv;
            }
        }
    }
}

// ---------------------------------------------------------------------------
// Device-wide barrier for the persistent scan (64 co-resident blocks).
// Release: fence + atomic arrive.  Acquire: spin (s_sleep) + fence.
// ---------------------------------------------------------------------------
__device__ __forceinline__ void grid_sync(unsigned* counter, unsigned target) {
    __syncthreads();
    __threadfence();    // make this block's h stores visible at device scope
    if (threadIdx.x == 0) {
        atomicAdd(counter, 1u);
        while (__hip_atomic_load(counter, __ATOMIC_ACQUIRE,
                                 __HIP_MEMORY_SCOPE_AGENT) < target) {
            __builtin_amdgcn_s_sleep(2);
        }
    }
    __syncthreads();
    __threadfence();    // invalidate stale cached h before next step's loads
}

// ---------------------------------------------------------------------------
// Persistent recurrent scan: runs all T steps of one layer.
//   h_{t+1} = tanh(xw[:,t,:] + h_t @ Whh^T)
// Grid = (HH/32, BB/32) = (32, 2) = 64 blocks; block = 64 threads = 2 waves.
// Each block owns a 32-column slice of Whh, staged once into LDS in WMMA
// B-operand order (lane-contiguous 32B -> conflict-minimal ds_load_b128).
// Wave tile = 16M x 32N (2 accumulators), K = 1024 from LDS.
// ---------------------------------------------------------------------------
__global__ __launch_bounds__(64)
void rnn_scan(const bf16* __restrict__ Whh,    // [H,H] bf16 (N,K)
              const float* __restrict__ xw,    // [B,T,H] f32 (bias included)
              float* __restrict__ y_f32,       // nullable [B,T,H]
              bf16*  __restrict__ y_bf,        // nullable [B,T,H]
              bf16*  __restrict__ hb0,         // [B,H] ping (zero-init)
              bf16*  __restrict__ hb1,         // [B,H] pong
              float* __restrict__ hlast,       // [B,H] final hidden state
              unsigned* __restrict__ counter)  // zero-init barrier counter
{
    const int lane = threadIdx.x & 31;
    const int wave = threadIdx.x >> 5;        // 0..1 along M
    const int mrow = lane & 15;
    const int ksel = lane >> 4;
    const int waveM = blockIdx.y * 32 + wave * 16;
    const int nBase = blockIdx.x * 32;        // 32 N-columns per block
    const size_t ldy = (size_t)TT * HH;

    // ------ stage Whh slice into LDS, pre-swizzled to B-operand layout -----
    // tile (kt, nt): 16N x 32K, 1 KB; lane's 16 halves at tile + lane*16.
    __shared__ bf16 sB[64 * 512];             // 64 tiles * 1KB = 64 KB
    for (int kt = wave; kt < 32; kt += 2) {
#pragma unroll
        for (int nt = 0; nt < 2; ++nt) {
            const bf16* src =
                Whh + (size_t)(nBase + nt * 16 + mrow) * HH + kt * 32 + ksel * 8;
            bf16* dst = sB + (kt * 2 + nt) * 512 + lane * 16;
            *(v8bf*)dst       = *(const v8bf*)src;
            *(v8bf*)(dst + 8) = *(const v8bf*)(src + 16);
        }
    }
    __syncthreads();

    const unsigned nBlocks = gridDim.x * gridDim.y;   // 64

    for (int t = 0; t < TT; ++t) {
        const bf16* hp = (t & 1) ? hb1 : hb0;
        bf16*       hn = (t & 1) ? hb0 : hb1;
        const bf16* Ab = hp + (size_t)(waveM + mrow) * HH + ksel * 8;

        v8f acc[2] = {};
        for (int kt = 0; kt < 32; ++kt) {
            ABf a;
            a.h[0] = *(const v8bf*)(Ab + kt * 32);
            a.h[1] = *(const v8bf*)(Ab + kt * 32 + 16);
#pragma unroll
            for (int nt = 0; nt < 2; ++nt) {
                const bf16* bs = sB + (kt * 2 + nt) * 512 + lane * 16;
                ABf bm;
                bm.h[0] = *(const v8bf*)bs;
                bm.h[1] = *(const v8bf*)(bs + 8);
                acc[nt] = __builtin_amdgcn_wmma_f32_16x16x32_bf16(
                    false, a.v, false, bm.v, (short)0, acc[nt], false, false);
            }
        }

        const float* xwt = xw + (size_t)t * HH;
#pragma unroll
        for (int nt = 0; nt < 2; ++nt) {
            const int col = nBase + nt * 16 + (lane & 15);
#pragma unroll
            for (int r = 0; r < 8; ++r) {
                const int row = waveM + ksel * 8 + r;
                const float v = tanhf(acc[nt][r] + xwt[(size_t)row * ldy + col]);
                hn[(size_t)row * HH + col] = (bf16)v;
                if (y_f32) (y_f32 + (size_t)t * HH)[(size_t)row * ldy + col] = v;
                if (y_bf)  (y_bf  + (size_t)t * HH)[(size_t)row * ldy + col] = (bf16)v;
                if (t == TT - 1) hlast[(size_t)row * HH + col] = v;
            }
        }

        if (t + 1 < TT) grid_sync(counter, (unsigned)(t + 1) * nBlocks);
    }
}

// ---------------------------------------------------------------------------
// Host launch
// ---------------------------------------------------------------------------
extern "C" void kernel_launch(void* const* d_in, const int* in_sizes, int n_in,
                              void* d_out, int out_size, void* d_ws, size_t ws_size,
                              hipStream_t stream) {
    const float* X    = (const float*)d_in[0];
    const float* Wih0 = (const float*)d_in[1];
    const float* bih0 = (const float*)d_in[2];
    const float* Whh0 = (const float*)d_in[3];
    const float* bhh0 = (const float*)d_in[4];
    const float* Wih1 = (const float*)d_in[5];
    const float* bih1 = (const float*)d_in[6];
    const float* Whh1 = (const float*)d_in[7];
    const float* bhh1 = (const float*)d_in[8];

    float* out_y = (float*)d_out;                       // [B,T,H]
    float* out_h = out_y + (size_t)BB * TT * HH;        // [2,B,H]

    const size_t actN = (size_t)BB * TT * HH;           // 33.5M elems
    const size_t wN   = (size_t)HH * HH;                // 1M elems

    // Workspace carve-up (chunks are power-of-2 sized -> aligned).
    char* p = (char*)d_ws;
    bf16* act_bf = (bf16*)p;  p += actN * sizeof(bf16);   // X bf16, later layer-0 out bf16
    float* xw    = (float*)p; p += actN * sizeof(float);  // per-layer input projection
    bf16* wih0b  = (bf16*)p;  p += wN * sizeof(bf16);
    bf16* whh0b  = (bf16*)p;  p += wN * sizeof(bf16);
    bf16* wih1b  = (bf16*)p;  p += wN * sizeof(bf16);
    bf16* whh1b  = (bf16*)p;  p += wN * sizeof(bf16);
    float* bsum0 = (float*)p; p += HH * sizeof(float);
    float* bsum1 = (float*)p; p += HH * sizeof(float);
    bf16* hbf0   = (bf16*)p;  p += (size_t)BB * HH * sizeof(bf16);
    bf16* hbf1   = (bf16*)p;  p += (size_t)BB * HH * sizeof(bf16);
    float* hf32  = (float*)p; p += (size_t)BB * HH * sizeof(float);
    unsigned* counter = (unsigned*)p; p += 256;
    (void)ws_size; (void)in_sizes; (void)n_in; (void)out_size;

    // --- conversions ---
    cvt_f32_to_bf16<<<8192, 256, 0, stream>>>(X, act_bf, (int)actN);
    cvt_f32_to_bf16<<<2048, 256, 0, stream>>>(Wih0, wih0b, (int)wN);
    cvt_f32_to_bf16<<<2048, 256, 0, stream>>>(Whh0, whh0b, (int)wN);
    cvt_f32_to_bf16<<<2048, 256, 0, stream>>>(Wih1, wih1b, (int)wN);
    cvt_f32_to_bf16<<<2048, 256, 0, stream>>>(Whh1, whh1b, (int)wN);
    bias_sum<<<HH / 256, 256, 0, stream>>>(bih0, bhh0, bsum0, HH);
    bias_sum<<<HH / 256, 256, 0, stream>>>(bih1, bhh1, bsum1, HH);

    const int M = BB * TT;  // 32768
    dim3 gemmGrid(HH / 256, M / 64);
    dim3 scanGrid(HH / 32, BB / 32);    // (32, 2) = 64 blocks

    // ---------------- layer 0 ----------------
    gemm_xw<<<gemmGrid, 256, 0, stream>>>(act_bf, wih0b, bsum0, xw, M, HH, HH);
    hipMemsetAsync(hbf0, 0, (size_t)BB * HH * sizeof(bf16), stream);
    hipMemsetAsync(counter, 0, sizeof(unsigned), stream);
    rnn_scan<<<scanGrid, 64, 0, stream>>>(whh0b, xw,
                                          /*y_f32=*/nullptr, /*y_bf=*/act_bf,
                                          hbf0, hbf1, hf32, counter);
    hipMemcpyAsync(out_h, hf32, (size_t)BB * HH * sizeof(float),
                   hipMemcpyDeviceToDevice, stream);

    // ---------------- layer 1 ----------------
    gemm_xw<<<gemmGrid, 256, 0, stream>>>(act_bf, wih1b, bsum1, xw, M, HH, HH);
    hipMemsetAsync(hbf0, 0, (size_t)BB * HH * sizeof(bf16), stream);
    hipMemsetAsync(counter, 0, sizeof(unsigned), stream);
    rnn_scan<<<scanGrid, 64, 0, stream>>>(whh1b, xw,
                                          /*y_f32=*/out_y, /*y_bf=*/nullptr,
                                          hbf0, hbf1, hf32, counter);
    hipMemcpyAsync(out_h + (size_t)BB * HH, hf32,
                   (size_t)BB * HH * sizeof(float),
                   hipMemcpyDeviceToDevice, stream);
}
